// MoELayer_3728031613100
// MI455X (gfx1250) — matrix operand order repaired and checked by reading, not compiled, not run on hardware
//
#include <hip/hip_runtime.h>
#include <cmath>

// ---------------------------------------------------------------------------
// MoE (gate -> top-2 dispatch -> SwiGLU experts -> weighted combine), gfx1250.
// Heavy GEMMs: V_WMMA_F32_16X16X32_BF16 (wave32), A-tiles staged with
// GLOBAL_LOAD_ASYNC_TO_LDS_B128 double-buffering (ASYNCcnt pipelined).
// Shapes: N=4096 tokens, D=1024, F=4096, E=8, top-k=2, gate hidden H=512.
// ---------------------------------------------------------------------------

typedef __attribute__((ext_vector_type(16))) __bf16 v16bf;
typedef __attribute__((ext_vector_type(8)))  float  v8f;

#define N_TOK 4096
#define DDIM  1024
#define FDIM  4096
#define NEXP  8
#define TOPK  2
#define HGATE 512

union FragU { uint4 u[2]; v16bf v; };   // 32B: one WMMA A/B fragment per lane
union Pack4 { __bf16 b[4]; uint2 u; };
union Pack8 { __bf16 b[8]; uint4 u; };

static __device__ inline v8f wmma_bf16(v16bf a, v16bf b, v8f c) {
  // D = A(16x32) * B(32x16) + C(16x16 f32)
  return __builtin_amdgcn_wmma_f32_16x16x32_bf16(
      /*neg_a=*/false, a, /*neg_b=*/false, b,
      /*c_mod=*/(short)0, c, /*reuse_a=*/false, /*reuse_b=*/false);
}

// LDS byte offset = low 32 bits of generic shared pointer (LDS aperture).
static __device__ inline unsigned lds_off(const void* p) {
  return (unsigned)(uintptr_t)p;
}
// Async DMA: 16B global -> LDS, tracked by ASYNCcnt (bypasses VGPRs).
static __device__ inline void async_ld_b128(unsigned dst_lds, const void* src) {
  asm volatile("global_load_async_to_lds_b128 %0, %1, off"
               :: "v"(dst_lds), "v"(src)
               : "memory");
}
template <int N>
static __device__ inline void wait_async() {
  asm volatile("s_wait_asynccnt %0" :: "n"(N) : "memory");
}

// --------------------------- small utility kernels -------------------------

__global__ void k_init_small(int* counts, int* fill, float* ssum) {
  int t = threadIdx.x;
  if (t < NEXP) { counts[t] = 0; fill[t] = 0; ssum[t] = 0.f; }
}

__global__ void k_zero_out(float* out, int n) {
  int i = blockIdx.x * blockDim.x + threadIdx.x;
  if (i < n) out[i] = 0.f;
}

__global__ void k_cvt_x(const float* __restrict__ x, __bf16* __restrict__ xb) {
  int i = blockIdx.x * blockDim.x + threadIdx.x;   // one float4 per thread
  float4 v = reinterpret_cast<const float4*>(x)[i];
  Pack4 p;
  p.b[0] = (__bf16)v.x; p.b[1] = (__bf16)v.y;
  p.b[2] = (__bf16)v.z; p.b[3] = (__bf16)v.w;
  reinterpret_cast<uint2*>(xb)[i] = p.u;
}

// --------------------------- gate layer 1 (WMMA) ---------------------------
// hidden[N,512] = relu(xb[N,1024] @ gw1[1024,512] + gb1)   (bf16 out)
// Block tile 128x128; 8 waves (4Mx2N), wave tile 32x64 = 2x4 WMMA tiles.
// A: async double-buffered; B: fp32 prefetched to regs, cvt+store per iter.

__launch_bounds__(256)
__global__ void k_gate1(const __bf16* __restrict__ xb, const float* __restrict__ gw1,
                        const float* __restrict__ gb1, __bf16* __restrict__ hidden) {
  __shared__ alignas(16) __bf16 As[2][128 * 40];  // [row][k], pitch 40 (80B)
  __shared__ alignas(16) __bf16 Bs[128 * 40];     // [col][k]
  const int tid  = threadIdx.x;
  const int wave = tid >> 5, lane = tid & 31;
  const int wm = wave >> 1, wn = wave & 1;
  const int rlo = lane & 15, hsel = lane >> 4;
  const int m0 = blockIdx.y * 128, n0 = blockIdx.x * 128;
  const int NK = DDIM / 32;

  v8f acc[2][4];
  for (int a = 0; a < 2; ++a)
    for (int b = 0; b < 4; ++b)
      for (int g = 0; g < 8; ++g) acc[a][b][g] = 0.f;

  const int ar0 = tid >> 2;            // A-tile row (pass 0), +64 for pass 1
  const int ac8 = (tid & 3) * 8;       // A-tile k sub-chunk

  auto issueA = [&](int k0, int buf) {
#pragma unroll
    for (int it = 0; it < 2; ++it) {
      int r = ar0 + it * 64;
      async_ld_b128(lds_off(&As[buf][r * 40 + ac8]),
                    xb + (size_t)(m0 + r) * DDIM + k0 + ac8);
    }
  };
  float breg[16];
  auto loadB = [&](int k0) {
#pragma unroll
    for (int i = 0; i < 16; ++i) {
      int idx = tid + i * 256;
      int k = idx >> 7, c = idx & 127;
      breg[i] = gw1[(size_t)(k0 + k) * HGATE + n0 + c];
    }
  };

  issueA(0, 0);
  loadB(0);

  for (int kt = 0; kt < NK; ++kt) {
    __syncthreads();                           // all waves done with compute(kt-1)
    // cvt + store B(kt) (regs were loaded last iteration)
#pragma unroll
    for (int i = 0; i < 16; ++i) {
      int idx = tid + i * 256;
      int k = idx >> 7, c = idx & 127;
      Bs[c * 40 + k] = (__bf16)breg[i];
    }
    if (kt + 1 < NK) {
      issueA((kt + 1) * 32, (kt + 1) & 1);     // prefetch next A tile (2 asyncs)
      loadB((kt + 1) * 32);                    // prefetch next B tile to regs
      wait_async<2>();                         // tile kt landed (in-order)
    } else {
      wait_async<0>();
    }
    __syncthreads();                           // tile kt visible to all waves
    const __bf16* Ab = As[kt & 1];
    v16bf af[2], bfr[4];
#pragma unroll
    for (int mi = 0; mi < 2; ++mi) {
      int row = wm * 32 + mi * 16 + rlo;
      int kf  = hsel * 8;                 // lanes 0-15: K 0..7/16..23; 16-31: 8..15/24..31
      FragU f;
      f.u[0] = *reinterpret_cast<const uint4*>(&Ab[row * 40 + kf]);
      f.u[1] = *reinterpret_cast<const uint4*>(&Ab[row * 40 + kf + 16]);
      af[mi] = f.v;
    }
#pragma unroll
    for (int ni = 0; ni < 4; ++ni) {
      int col = wn * 64 + ni * 16 + rlo;
      int kf  = hsel * 8;                 // lanes 0-15: K 0..7/16..23; 16-31: 8..15/24..31
      FragU f;
      f.u[0] = *reinterpret_cast<const uint4*>(&Bs[col * 40 + kf]);
      f.u[1] = *reinterpret_cast<const uint4*>(&Bs[col * 40 + kf + 16]);
      bfr[ni] = f.v;
    }
#pragma unroll
    for (int mi = 0; mi < 2; ++mi)
#pragma unroll
      for (int ni = 0; ni < 4; ++ni)
        acc[mi][ni] = wmma_bf16(af[mi], bfr[ni], acc[mi][ni]);
  }
  // epilogue: +bias, relu, bf16 store
#pragma unroll
  for (int mi = 0; mi < 2; ++mi) {
    int row0 = m0 + wm * 32 + mi * 16 + hsel * 8;
#pragma unroll
    for (int ni = 0; ni < 4; ++ni) {
      int col  = n0 + wn * 64 + ni * 16 + rlo;
      float bias = gb1[col];
#pragma unroll
      for (int g = 0; g < 8; ++g) {
        float v = acc[mi][ni][g] + bias;
        v = v > 0.f ? v : 0.f;
        hidden[(size_t)(row0 + g) * HGATE + col] = (__bf16)v;
      }
    }
  }
}

// ------------------ gate layer 2 + softmax + top-2 + stats -----------------
// One wave per token (8 tokens per 256-thread block).

__launch_bounds__(256)
__global__ void k_gate2(const __bf16* __restrict__ hidden, const float* __restrict__ gw2,
                        int* __restrict__ ti, float* __restrict__ tw,
                        int* __restrict__ counts, float* __restrict__ ssum) {
  const int lane = threadIdx.x & 31;
  const int t = blockIdx.x * 8 + (threadIdx.x >> 5);
  float acc[NEXP];
#pragma unroll
  for (int e = 0; e < NEXP; ++e) acc[e] = 0.f;
#pragma unroll
  for (int c = 0; c < 2; ++c) {
    Pack8 p;
    p.u = *reinterpret_cast<const uint4*>(hidden + (size_t)t * HGATE + lane * 16 + c * 8);
#pragma unroll
    for (int j = 0; j < 8; ++j) {
      float hv = (float)p.b[j];
      int idx = lane * 16 + c * 8 + j;
#pragma unroll
      for (int e = 0; e < NEXP; ++e) acc[e] += hv * gw2[idx * NEXP + e];
    }
  }
#pragma unroll
  for (int off = 16; off >= 1; off >>= 1)
#pragma unroll
    for (int e = 0; e < NEXP; ++e) acc[e] += __shfl_xor(acc[e], off, 32);

  if (lane == 0) {
    float mx = acc[0];
    for (int e = 1; e < NEXP; ++e) mx = fmaxf(mx, acc[e]);
    float p[NEXP], s = 0.f;
    for (int e = 0; e < NEXP; ++e) { p[e] = __expf(acc[e] - mx); s += p[e]; }
    float inv = 1.f / s;
    int i0 = 0; float v0 = -1.f;
    for (int e = 0; e < NEXP; ++e) { p[e] *= inv; if (p[e] > v0) { v0 = p[e]; i0 = e; } }
    int i1 = 0; float v1 = -1.f;
    for (int e = 0; e < NEXP; ++e) { if (e != i0 && p[e] > v1) { v1 = p[e]; i1 = e; } }
    for (int e = 0; e < NEXP; ++e) unsafeAtomicAdd(&ssum[e], p[e]);  // balance-loss mean
    // renormalized top-2 weights: softmax over the two top scores
    float e0 = __expf(v0), e1 = __expf(v1);
    float w0 = e0 / (e0 + e1);
    ti[t * 2 + 0] = i0; ti[t * 2 + 1] = i1;
    tw[t * 2 + 0] = w0; tw[t * 2 + 1] = 1.f - w0;
    atomicAdd(&counts[i0], 1);
    atomicAdd(&counts[i1], 1);
  }
}

// segment offsets + balance loss finalize
__global__ void k_scan(const int* __restrict__ counts, int* __restrict__ offs,
                       const float* __restrict__ ssum, float* __restrict__ loss_out) {
  if (threadIdx.x == 0) {
    int a = 0;
    for (int e = 0; e < NEXP; ++e) { offs[e] = a; a += counts[e]; }
    offs[NEXP] = a;
    float L = 0.f;
    for (int e = 0; e < NEXP; ++e) {
      float m = ssum[e] / (float)N_TOK;
      L += m * logf(m + 1e-8f);
    }
    *loss_out = (float)NEXP * L;
  }
}

__global__ void k_scatter(const int* __restrict__ ti, const float* __restrict__ tw,
                          const int* __restrict__ offs, int* __restrict__ fill,
                          int* __restrict__ tok_id, float* __restrict__ tok_w) {
  int t = blockIdx.x * blockDim.x + threadIdx.x;
  if (t >= N_TOK) return;
#pragma unroll
  for (int k = 0; k < TOPK; ++k) {
    int e = ti[t * 2 + k];
    int slot = atomicAdd(&fill[e], 1);
    int pos = offs[e] + slot;
    tok_id[pos] = t;
    tok_w[pos]  = tw[t * 2 + k];
  }
}

// gather token rows into per-expert-contiguous bf16 activation buffer
__global__ void k_gather(const __bf16* __restrict__ xb, const int* __restrict__ tok_id,
                         __bf16* __restrict__ xg) {
  int pos = blockIdx.x;
  int t = tok_id[pos];
  const uint4* s = reinterpret_cast<const uint4*>(xb + (size_t)t * DDIM);
  uint4* d = reinterpret_cast<uint4*>(xg + (size_t)pos * DDIM);
  d[threadIdx.x] = s[threadIdx.x];            // 128 threads * 16B = 2KB row
}

// --------------- expert GEMM 1+2 fused: h = silu(x@W1) * (x@W2) ------------
// Block tile 128(M)x64(F), K=D loop; 8 waves 4Mx2N, wave tile 32x32 = 2x2.
// Out-of-segment rows are clamped (outputs dropped by the store guard) so the
// async-issue count per wave stays exact for the ASYNCcnt pipeline.

__launch_bounds__(256)
__global__ void k_expert_h(const __bf16* __restrict__ xg, const float* __restrict__ W1,
                           const float* __restrict__ W2, const int* __restrict__ offs,
                           __bf16* __restrict__ hbuf) {
  const int e = blockIdx.z;
  const int seg0 = offs[e], seg1 = offs[e + 1];
  const int rows = seg1 - seg0;
  const int m0 = blockIdx.y * 128;
  if (m0 >= rows) return;                      // early-exit on real segment size
  const int f0 = blockIdx.x * 64;

  __shared__ alignas(16) __bf16 As[2][128 * 40];
  __shared__ alignas(16) __bf16 B1s[64 * 40];
  __shared__ alignas(16) __bf16 B2s[64 * 40];

  const int tid = threadIdx.x;
  const int wave = tid >> 5, lane = tid & 31;
  const int wm = wave >> 1, wn = wave & 1;
  const int rlo = lane & 15, hsel = lane >> 4;
  const float* W1e = W1 + (size_t)e * DDIM * FDIM;
  const float* W2e = W2 + (size_t)e * DDIM * FDIM;
  const int NK = DDIM / 32;

  v8f a1[2][2], a2[2][2];
  for (int a = 0; a < 2; ++a)
    for (int b = 0; b < 2; ++b)
      for (int g = 0; g < 8; ++g) { a1[a][b][g] = 0.f; a2[a][b][g] = 0.f; }

  const int ar0 = tid >> 2;
  const int ac8 = (tid & 3) * 8;
  auto issueA = [&](int k0, int buf) {
#pragma unroll
    for (int it = 0; it < 2; ++it) {
      int r  = ar0 + it * 64;
      int rg = min(m0 + r, rows - 1);          // clamp: dead rows read a valid row
      async_ld_b128(lds_off(&As[buf][r * 40 + ac8]),
                    xg + (size_t)(seg0 + rg) * DDIM + k0 + ac8);
    }
  };
  float b1r[8], b2r[8];
  auto loadB = [&](int k0) {
#pragma unroll
    for (int i = 0; i < 8; ++i) {
      int idx = tid + i * 256;
      int k = idx >> 6, c = idx & 63;
      size_t goff = (size_t)(k0 + k) * FDIM + f0 + c;
      b1r[i] = W1e[goff];
      b2r[i] = W2e[goff];
    }
  };

  issueA(0, 0);
  loadB(0);

  for (int kt = 0; kt < NK; ++kt) {
    __syncthreads();
#pragma unroll
    for (int i = 0; i < 8; ++i) {
      int idx = tid + i * 256;
      int k = idx >> 6, c = idx & 63;
      B1s[c * 40 + k] = (__bf16)b1r[i];
      B2s[c * 40 + k] = (__bf16)b2r[i];
    }
    if (kt + 1 < NK) {
      issueA((kt + 1) * 32, (kt + 1) & 1);
      loadB((kt + 1) * 32);
      wait_async<2>();
    } else {
      wait_async<0>();
    }
    __syncthreads();
    const __bf16* Ab = As[kt & 1];
    v16bf af[2], b1f[2], b2f[2];
#pragma unroll
    for (int mi = 0; mi < 2; ++mi) {
      int row = wm * 32 + mi * 16 + rlo;
      int kf  = hsel * 8;
      FragU f;
      f.u[0] = *reinterpret_cast<const uint4*>(&Ab[row * 40 + kf]);
      f.u[1] = *reinterpret_cast<const uint4*>(&Ab[row * 40 + kf + 16]);
      af[mi] = f.v;
    }
#pragma unroll
    for (int ni = 0; ni < 2; ++ni) {
      int col = wn * 32 + ni * 16 + rlo;
      int kf  = hsel * 8;
      FragU f;
      f.u[0] = *reinterpret_cast<const uint4*>(&B1s[col * 40 + kf]);
      f.u[1] = *reinterpret_cast<const uint4*>(&B1s[col * 40 + kf + 16]);
      b1f[ni] = f.v;
      f.u[0] = *reinterpret_cast<const uint4*>(&B2s[col * 40 + kf]);
      f.u[1] = *reinterpret_cast<const uint4*>(&B2s[col * 40 + kf + 16]);
      b2f[ni] = f.v;
    }
#pragma unroll
    for (int mi = 0; mi < 2; ++mi)
#pragma unroll
      for (int ni = 0; ni < 2; ++ni) {
        a1[mi][ni] = wmma_bf16(af[mi], b1f[ni], a1[mi][ni]);
        a2[mi][ni] = wmma_bf16(af[mi], b2f[ni], a2[mi][ni]);
      }
  }
#pragma unroll
  for (int mi = 0; mi < 2; ++mi) {
    int grow0 = m0 + wm * 32 + mi * 16 + hsel * 8;
#pragma unroll
    for (int ni = 0; ni < 2; ++ni) {
      int fc = f0 + wn * 32 + ni * 16 + rlo;
#pragma unroll
      for (int g = 0; g < 8; ++g) {
        int grow = grow0 + g;
        if (grow < rows) {
          float g1 = a1[mi][ni][g];
          float g2 = a2[mi][ni][g];
          float s = g1 / (1.f + __expf(-g1));   // silu
          hbuf[(size_t)(seg0 + grow) * FDIM + fc] = (__bf16)(s * g2);
        }
      }
    }
  }
}

// ------------- expert GEMM 3: out += top_w * (h @ W3), scattered -----------

__launch_bounds__(256)
__global__ void k_expert_out(const __bf16* __restrict__ hbuf, const float* __restrict__ W3,
                             const int* __restrict__ offs, const int* __restrict__ tok_id,
                             const float* __restrict__ tok_w, float* __restrict__ out) {
  const int e = blockIdx.z;
  const int seg0 = offs[e], seg1 = offs[e + 1];
  const int rows = seg1 - seg0;
  const int m0 = blockIdx.y * 128;
  if (m0 >= rows) return;
  const int d0 = blockIdx.x * 64;

  __shared__ alignas(16) __bf16 As[2][128 * 40];
  __shared__ alignas(16) __bf16 Bs[64 * 40];

  const int tid = threadIdx.x;
  const int wave = tid >> 5, lane = tid & 31;
  const int wm = wave >> 1, wn = wave & 1;
  const int rlo = lane & 15, hsel = lane >> 4;
  const float* W3e = W3 + (size_t)e * FDIM * DDIM;
  const int NK = FDIM / 32;

  v8f acc[2][2];
  for (int a = 0; a < 2; ++a)
    for (int b = 0; b < 2; ++b)
      for (int g = 0; g < 8; ++g) acc[a][b][g] = 0.f;

  const int ar0 = tid >> 2;
  const int ac8 = (tid & 3) * 8;
  auto issueA = [&](int k0, int buf) {
#pragma unroll
    for (int it = 0; it < 2; ++it) {
      int r  = ar0 + it * 64;
      int rg = min(m0 + r, rows - 1);
      async_ld_b128(lds_off(&As[buf][r * 40 + ac8]),
                    hbuf + (size_t)(seg0 + rg) * FDIM + k0 + ac8);
    }
  };
  float breg[8];
  auto loadB = [&](int k0) {
#pragma unroll
    for (int i = 0; i < 8; ++i) {
      int idx = tid + i * 256;
      int k = idx >> 6, c = idx & 63;
      breg[i] = W3e[(size_t)(k0 + k) * DDIM + d0 + c];
    }
  };

  issueA(0, 0);
  loadB(0);

  for (int kt = 0; kt < NK; ++kt) {
    __syncthreads();
#pragma unroll
    for (int i = 0; i < 8; ++i) {
      int idx = tid + i * 256;
      int k = idx >> 6, c = idx & 63;
      Bs[c * 40 + k] = (__bf16)breg[i];
    }
    if (kt + 1 < NK) {
      issueA((kt + 1) * 32, (kt + 1) & 1);
      loadB((kt + 1) * 32);
      wait_async<2>();
    } else {
      wait_async<0>();
    }
    __syncthreads();
    const __bf16* Ab = As[kt & 1];
    v16bf af[2], bfr[2];
#pragma unroll
    for (int mi = 0; mi < 2; ++mi) {
      int row = wm * 32 + mi * 16 + rlo;
      int kf  = hsel * 8;
      FragU f;
      f.u[0] = *reinterpret_cast<const uint4*>(&Ab[row * 40 + kf]);
      f.u[1] = *reinterpret_cast<const uint4*>(&Ab[row * 40 + kf + 16]);
      af[mi] = f.v;
    }
#pragma unroll
    for (int ni = 0; ni < 2; ++ni) {
      int col = wn * 32 + ni * 16 + rlo;
      int kf  = hsel * 8;
      FragU f;
      f.u[0] = *reinterpret_cast<const uint4*>(&Bs[col * 40 + kf]);
      f.u[1] = *reinterpret_cast<const uint4*>(&Bs[col * 40 + kf + 16]);
      bfr[ni] = f.v;
    }
#pragma unroll
    for (int mi = 0; mi < 2; ++mi)
#pragma unroll
      for (int ni = 0; ni < 2; ++ni)
        acc[mi][ni] = wmma_bf16(af[mi], bfr[ni], acc[mi][ni]);
  }
#pragma unroll
  for (int mi = 0; mi < 2; ++mi) {
    int grow0 = m0 + wm * 32 + mi * 16 + hsel * 8;
#pragma unroll
    for (int ni = 0; ni < 2; ++ni) {
      int dc = d0 + wn * 32 + ni * 16 + rlo;
#pragma unroll
      for (int g = 0; g < 8; ++g) {
        int grow = grow0 + g;
        if (grow < rows) {
          int pos = seg0 + grow;
          int t = tok_id[pos];
          float w = tok_w[pos];
          unsafeAtomicAdd(&out[(size_t)t * DDIM + dc], acc[mi][ni][g] * w);
        }
      }
    }
  }
}

// ------------------------------- launcher ----------------------------------

extern "C" void kernel_launch(void* const* d_in, const int* in_sizes, int n_in,
                              void* d_out, int out_size, void* d_ws, size_t ws_size,
                              hipStream_t stream) {
  (void)in_sizes; (void)n_in; (void)out_size; (void)ws_size;
  const float* x   = (const float*)d_in[0];
  const float* gw1 = (const float*)d_in[1];
  const float* gb1 = (const float*)d_in[2];
  const float* gw2 = (const float*)d_in[3];
  const float* W1  = (const float*)d_in[4];
  const float* W2  = (const float*)d_in[5];
  const float* W3  = (const float*)d_in[6];
  float* out = (float*)d_out;

  // workspace carve-out (~96.5 MB total)
  char* ws = (char*)d_ws;
  size_t off = 0;
  auto alloc = [&](size_t bytes) -> char* {
    char* p = ws + off;
    off = (off + bytes + 255) & ~(size_t)255;
    return p;
  };
  __bf16* xb     = (__bf16*)alloc((size_t)N_TOK * DDIM * 2);        // 8 MB
  __bf16* hidden = (__bf16*)alloc((size_t)N_TOK * HGATE * 2);       // 4 MB
  int*    ti     = (int*)  alloc((size_t)N_TOK * TOPK * 4);
  float*  tw     = (float*)alloc((size_t)N_TOK * TOPK * 4);
  int*    tok_id = (int*)  alloc((size_t)N_TOK * TOPK * 4);
  float*  tok_w  = (float*)alloc((size_t)N_TOK * TOPK * 4);
  int*    counts = (int*)  alloc(NEXP * 4);
  int*    fill   = (int*)  alloc(NEXP * 4);
  float*  ssum   = (float*)alloc(NEXP * 4);
  int*    offs   = (int*)  alloc((NEXP + 1) * 4);
  __bf16* xg     = (__bf16*)alloc((size_t)N_TOK * TOPK * DDIM * 2); // 16 MB
  __bf16* hbuf   = (__bf16*)alloc((size_t)N_TOK * TOPK * FDIM * 2); // 64 MB

  k_init_small<<<1, 32, 0, stream>>>(counts, fill, ssum);
  const int ntot = N_TOK * DDIM + 1;
  k_zero_out<<<(ntot + 255) / 256, 256, 0, stream>>>(out, ntot);
  k_cvt_x<<<(N_TOK * DDIM / 4) / 256, 256, 0, stream>>>(x, xb);
  k_gate1<<<dim3(HGATE / 128, N_TOK / 128), 256, 0, stream>>>(xb, gw1, gb1, hidden);
  k_gate2<<<N_TOK / 8, 256, 0, stream>>>(hidden, gw2, ti, tw, counts, ssum);
  k_scan<<<1, 32, 0, stream>>>(counts, offs, ssum, out + (size_t)N_TOK * DDIM);
  k_scatter<<<(N_TOK + 255) / 256, 256, 0, stream>>>(ti, tw, offs, fill, tok_id, tok_w);
  k_gather<<<N_TOK * TOPK, 128, 0, stream>>>(xb, tok_id, xg);
  k_expert_h<<<dim3(FDIM / 64, (N_TOK * TOPK) / 128, NEXP), 256, 0, stream>>>(
      xg, W1, W2, offs, hbuf);
  k_expert_out<<<dim3(DDIM / 64, (N_TOK * TOPK) / 128, NEXP), 256, 0, stream>>>(
      hbuf, W3, offs, tok_id, tok_w, out);
}
